// ConcatCLS_HyperlinkPredictionHead_29832842838777
// MI455X (gfx1250) — matrix operand order repaired and checked by reading, not compiled, not run on hardware
//
#include <hip/hip_runtime.h>
#include <hip/hip_bf16.h>

typedef __bf16 bf16_t;
typedef __attribute__((ext_vector_type(16))) __bf16 v16bf;
typedef __attribute__((ext_vector_type(8)))  __bf16 v8bf;
typedef __attribute__((ext_vector_type(8)))  float  v8f;

#define TT     2048
#define HDIM   768
#define IN_F   1536
#define HEAD_N 10002
#define T1K    384
#define T1N    30000
#define T2K    96
#define T2N    60000
#define CUT0   10000
#define CUT1   40000

__device__ __forceinline__ int dmin(int a, int b) { return a < b ? a : b; }

__device__ __forceinline__ v8f wmma_bf16(v16bf a, v16bf b, v8f c) {
  // v_wmma_f32_16x16x32_bf16: (neg_a, A, neg_b, B, c_mod, C, reuse_a, reuse_b)
  return __builtin_amdgcn_wmma_f32_16x16x32_bf16(false, a, false, b, (short)0, c, false, false);
}

// A fragment: lane holds row (lane&15); K contiguous [kk+16*hi .. +15]
__device__ __forceinline__ v16bf load_a_frag(const bf16_t* __restrict__ p) {
  v8bf lo = *(const v8bf*)p;
  v8bf hi = *(const v8bf*)(p + 8);
  return __builtin_shufflevector(lo, hi, 0,1,2,3,4,5,6,7,8,9,10,11,12,13,14,15);
}

// ---------------- fp32 -> bf16 convert ----------------
__global__ void f2bf_kernel(const float* __restrict__ in, bf16_t* __restrict__ out, int n) {
  int i = blockIdx.x * blockDim.x + threadIdx.x;
  int stride = gridDim.x * blockDim.x;
  for (; i < n; i += stride) out[i] = (bf16_t)in[i];
}

// ---------------- ragged gather + concat -> bf16 emb [TT, IN_F] ----------------
__global__ void build_emb_kernel(const float* __restrict__ lhs, const float* __restrict__ cls,
                                 const int* __restrict__ eids, const int* __restrict__ tids,
                                 bf16_t* __restrict__ emb) {
  int t = blockIdx.x;
  int e = eids[t], k = tids[t];
  const float* a = cls + (size_t)e * HDIM;
  const float* b = lhs + ((size_t)e * 512 + k) * HDIM;
  bf16_t* dst = emb + (size_t)t * IN_F;
  for (int i = threadIdx.x; i < HDIM; i += blockDim.x) {
    dst[i]        = (bf16_t)a[i];
    dst[HDIM + i] = (bf16_t)b[i];
  }
}

// ---------------- gather-index precompute ----------------
__global__ void gind_kernel(const int* __restrict__ targets, int* __restrict__ gH,
                            int* __restrict__ g1, int* __restrict__ g2) {
  int r = blockIdx.x * blockDim.x + threadIdx.x;
  if (r >= TT) return;
  int t = targets[r];
  gH[r] = t < CUT0 ? t : (t < CUT1 ? CUT0 : CUT0 + 1);
  g1[r] = (t >= CUT0 && t < CUT1) ? (t - CUT0) : -1;
  g2[r] = (t >= CUT1) ? dmin(t - CUT1, T2N - 1) : -1;
}

// ---------------- projection GEMM: C_bf[M=TT,N] = A[TT,K] @ W[N,K]^T ----------------
__global__ void __launch_bounds__(32)
gemm_proj_kernel(const bf16_t* __restrict__ A, int K,
                 const bf16_t* __restrict__ W, int N, bf16_t* __restrict__ C) {
  const int lane = threadIdx.x, half = lane >> 4, l15 = lane & 15;
  const int rowBase = blockIdx.x * 32, colBase = blockIdx.y * 32;
  const bf16_t* a0p = A + (size_t)(rowBase + l15) * K + (half ? 16 : 0);
  const bf16_t* a1p = a0p + (size_t)16 * K;
  const bf16_t* b0p = W + (size_t)(colBase + l15) * K + (half ? 16 : 0);
  const bf16_t* b1p = b0p + (size_t)16 * K;
  v8f z = {};
  v8f acc00 = z, acc01 = z, acc10 = z, acc11 = z;
  for (int kk = 0; kk < K; kk += 32) {
    v16bf b0 = *(const v16bf*)(b0p + kk);
    v16bf b1 = *(const v16bf*)(b1p + kk);
    v16bf a0 = load_a_frag(a0p + kk);
    v16bf a1 = load_a_frag(a1p + kk);
    acc00 = wmma_bf16(a0, b0, acc00);
    acc01 = wmma_bf16(a0, b1, acc01);
    acc10 = wmma_bf16(a1, b0, acc10);
    acc11 = wmma_bf16(a1, b1, acc11);
  }
#pragma unroll
  for (int v = 0; v < 8; ++v) {
    int r0 = rowBase + v + 8 * half, r1 = r0 + 16;
    int c0 = colBase + l15, c1 = c0 + 16;
    C[(size_t)r0 * N + c0] = (bf16_t)acc00[v];
    C[(size_t)r0 * N + c1] = (bf16_t)acc01[v];
    C[(size_t)r1 * N + c0] = (bf16_t)acc10[v];
    C[(size_t)r1 * N + c1] = (bf16_t)acc11[v];
  }
}

// ---------------- fused logits GEMM + online logsumexp + target capture ----------------
// wave: 32 rows x (strip of 32-col blocks). partial[strip*TT + row] = (runMax, runSum)
// K compile-time; HOIST_A keeps the wave's whole A panel in VGPRs (viable for K=96).
template <int K, bool HOIST_A>
__global__ void __launch_bounds__(32)
fused_lse_kernel(const bf16_t* __restrict__ A,
                 const bf16_t* __restrict__ W, int N,
                 const float* __restrict__ bias,
                 const int* __restrict__ gind,
                 float* __restrict__ tgt,
                 float2* __restrict__ partial,
                 int blocksPerStrip) {
  constexpr int KSTEPS = K / 32;
  const int lane = threadIdx.x, half = lane >> 4, l15 = lane & 15;
  const int rowBase = blockIdx.x * 32;
  const int strip = blockIdx.y;
  const int nBlocks = (N + 31) / 32;
  const int blk0 = strip * blocksPerStrip;
  const int blkEnd = dmin(blk0 + blocksPerStrip, nBlocks);

  int gi[2][8];
#pragma unroll
  for (int mt = 0; mt < 2; ++mt)
#pragma unroll
    for (int v = 0; v < 8; ++v)
      gi[mt][v] = gind[rowBase + 16 * mt + v + 8 * half];

  float runM[2][8], runS[2][8];
#pragma unroll
  for (int mt = 0; mt < 2; ++mt)
#pragma unroll
    for (int v = 0; v < 8; ++v) { runM[mt][v] = -1e30f; runS[mt][v] = 0.0f; }

  const bf16_t* a0p = A + (size_t)(rowBase + l15) * K + (half ? 16 : 0);
  const bf16_t* a1p = a0p + (size_t)16 * K;

  // Hoisted A panel (whole K) for small-K matrices: read A exactly once per wave.
  v16bf aF[HOIST_A ? KSTEPS : 1][2];
  if constexpr (HOIST_A) {
#pragma unroll
    for (int ks = 0; ks < KSTEPS; ++ks) {
      aF[ks][0] = load_a_frag(a0p + ks * 32);
      aF[ks][1] = load_a_frag(a1p + ks * 32);
    }
  }

  for (int blk = blk0; blk < blkEnd; ++blk) {
    const int colBase = blk * 32;
    const int c0 = colBase + l15, c1 = c0 + 16;
    const int n0 = dmin(c0, N - 1), n1 = dmin(c1, N - 1);
    const bool ok0 = c0 < N, ok1 = c1 < N;
    const bf16_t* b0p = W + (size_t)n0 * K + (half ? 16 : 0);
    const bf16_t* b1p = W + (size_t)n1 * K + (half ? 16 : 0);

    // Prefetch next column-block's B rows into cache (global_prefetch_b8)
    // while this block's ~KSTEPS*4 WMMAs execute.
    if (blk + 1 < blkEnd) {
      const int pn0 = dmin(colBase + 32 + l15, N - 1);
      const int pn1 = dmin(colBase + 48 + l15, N - 1);
      __builtin_prefetch(W + (size_t)pn0 * K, 0, 3);
      __builtin_prefetch(W + (size_t)pn1 * K, 0, 3);
    }

    v8f z = {};
    v8f acc00 = z, acc01 = z, acc10 = z, acc11 = z;
#pragma unroll
    for (int ks = 0; ks < KSTEPS; ++ks) {
      const int kk = ks * 32;
      v16bf b0 = *(const v16bf*)(b0p + kk);
      v16bf b1 = *(const v16bf*)(b1p + kk);
      v16bf a0, a1;
      if constexpr (HOIST_A) {
        a0 = aF[ks][0];
        a1 = aF[ks][1];
      } else {
        a0 = load_a_frag(a0p + kk);
        a1 = load_a_frag(a1p + kk);
      }
      acc00 = wmma_bf16(a0, b0, acc00);
      acc01 = wmma_bf16(a0, b1, acc01);
      acc10 = wmma_bf16(a1, b0, acc10);
      acc11 = wmma_bf16(a1, b1, acc11);
    }

    const float bb0 = bias ? bias[n0] : 0.0f;
    const float bb1 = bias ? bias[n1] : 0.0f;

#pragma unroll
    for (int mt = 0; mt < 2; ++mt) {
#pragma unroll
      for (int v = 0; v < 8; ++v) {
        float x0 = (mt == 0 ? acc00[v] : acc10[v]);
        float x1 = (mt == 0 ? acc01[v] : acc11[v]);
        x0 = ok0 ? x0 + bb0 : -1e30f;
        x1 = ok1 ? x1 + bb1 : -1e30f;
        const int row = rowBase + 16 * mt + v + 8 * half;
        if (ok0 && c0 == gi[mt][v]) tgt[row] = x0;
        if (ok1 && c1 == gi[mt][v]) tgt[row] = x1;
        // per-row reduction across 16 columns (lanes within same half)
        float M = fmaxf(x0, x1);
#pragma unroll
        for (int m = 1; m <= 8; m <<= 1) M = fmaxf(M, __shfl_xor(M, m, 32));
        float S = __expf(x0 - M) + __expf(x1 - M);
#pragma unroll
        for (int m = 1; m <= 8; m <<= 1) S += __shfl_xor(S, m, 32);
        // online merge
        float nm = fmaxf(runM[mt][v], M);
        runS[mt][v] = runS[mt][v] * __expf(runM[mt][v] - nm) + S * __expf(M - nm);
        runM[mt][v] = nm;
      }
    }
  }

  if (l15 == 0) {
#pragma unroll
    for (int mt = 0; mt < 2; ++mt)
#pragma unroll
      for (int v = 0; v < 8; ++v) {
        int row = rowBase + 16 * mt + v + 8 * half;
        partial[(size_t)strip * TT + row] = make_float2(runM[mt][v], runS[mt][v]);
      }
  }
}

// ---------------- final per-row merge + gather composition ----------------
__device__ __forceinline__ float merge_lse(const float2* p, int ns, int r) {
  float M = -1e30f, S = 0.0f;
  for (int i = 0; i < ns; ++i) {
    float2 q = p[(size_t)i * TT + r];
    float nm = fmaxf(M, q.x);
    S = S * __expf(M - nm) + q.y * __expf(q.x - nm);
    M = nm;
  }
  return M + __logf(S);
}

__global__ void reduce_out_kernel(const int* __restrict__ targets,
                                  const float* __restrict__ tgtH, const float* __restrict__ tgt1,
                                  const float* __restrict__ tgt2,
                                  const float2* __restrict__ pH, int sH,
                                  const float2* __restrict__ p1, int s1,
                                  const float2* __restrict__ p2, int s2,
                                  float* __restrict__ out) {
  int r = blockIdx.x * blockDim.x + threadIdx.x;
  if (r >= TT) return;
  float o = tgtH[r] - merge_lse(pH, sH, r);
  int t = targets[r];
  if (t >= CUT0 && t < CUT1) o += tgt1[r] - merge_lse(p1, s1, r);
  if (t >= CUT1)             o += tgt2[r] - merge_lse(p2, s2, r);
  out[r] = o;
}

__global__ void loss_kernel(const float* __restrict__ out, float* __restrict__ loss) {
  __shared__ float sm[256];
  float s = 0.0f;
  for (int i = threadIdx.x; i < TT; i += 256) s += out[i];
  sm[threadIdx.x] = s;
  __syncthreads();
  for (int w = 128; w > 0; w >>= 1) {
    if (threadIdx.x < w) sm[threadIdx.x] += sm[threadIdx.x + w];
    __syncthreads();
  }
  if (threadIdx.x == 0) loss[0] = -sm[0] / (float)TT;
}

// ---------------- host launch ----------------
static inline size_t alignup(size_t x) { return (x + 255) & ~(size_t)255; }

extern "C" void kernel_launch(void* const* d_in, const int* in_sizes, int n_in,
                              void* d_out, int out_size, void* d_ws, size_t ws_size,
                              hipStream_t stream) {
  const int*   targets = (const int*)d_in[0];
  const float* lhs     = (const float*)d_in[1];
  const float* cls     = (const float*)d_in[2];
  const int*   eids    = (const int*)d_in[3];
  const int*   tids    = (const int*)d_in[4];
  const float* head_w  = (const float*)d_in[5];
  const float* head_b  = (const float*)d_in[6];
  const float* t1p_w   = (const float*)d_in[7];
  const float* t1o_w   = (const float*)d_in[8];
  const float* t2p_w   = (const float*)d_in[9];
  const float* t2o_w   = (const float*)d_in[10];
  float* out = (float*)d_out;

  // strip configs
  const int nbH = (HEAD_N + 31) / 32, bpsH = 10, sH = (nbH + bpsH - 1) / bpsH; // 313 -> 32
  const int nb1 = (T1N + 31) / 32,   bps1 = 16, s1 = (nb1 + bps1 - 1) / bps1; // 938 -> 59
  const int nb2 = (T2N + 31) / 32,   bps2 = 16, s2 = (nb2 + bps2 - 1) / bps2; // 1875 -> 118

  // workspace carve-up
  char* ws = (char*)d_ws;
  size_t off = 0;
  auto carve = [&](size_t bytes) { char* p = ws + off; off = alignup(off + bytes); return p; };
  bf16_t* emb   = (bf16_t*)carve((size_t)TT * IN_F * 2);
  bf16_t* hWbf  = (bf16_t*)carve((size_t)HEAD_N * IN_F * 2);
  bf16_t* t1pbf = (bf16_t*)carve((size_t)T1K * IN_F * 2);
  bf16_t* t1obf = (bf16_t*)carve((size_t)T1N * T1K * 2);
  bf16_t* t2pbf = (bf16_t*)carve((size_t)T2K * IN_F * 2);
  bf16_t* t2obf = (bf16_t*)carve((size_t)T2N * T2K * 2);
  bf16_t* h1    = (bf16_t*)carve((size_t)TT * T1K * 2);
  bf16_t* h2    = (bf16_t*)carve((size_t)TT * T2K * 2);
  int* gH  = (int*)carve(TT * 4);
  int* g1  = (int*)carve(TT * 4);
  int* g2  = (int*)carve(TT * 4);
  float* tgtH = (float*)carve(TT * 4);
  float* tgt1 = (float*)carve(TT * 4);
  float* tgt2 = (float*)carve(TT * 4);
  float2* pH = (float2*)carve((size_t)sH * TT * 8);
  float2* p1 = (float2*)carve((size_t)s1 * TT * 8);
  float2* p2 = (float2*)carve((size_t)s2 * TT * 8);
  (void)ws_size; (void)in_sizes; (void)n_in; (void)out_size;

  // 1) weight converts fp32 -> bf16
  f2bf_kernel<<<2048, 256, 0, stream>>>(head_w, hWbf, HEAD_N * IN_F);
  f2bf_kernel<<<512,  256, 0, stream>>>(t1p_w,  t1pbf, T1K * IN_F);
  f2bf_kernel<<<2048, 256, 0, stream>>>(t1o_w,  t1obf, T1N * T1K);
  f2bf_kernel<<<256,  256, 0, stream>>>(t2p_w,  t2pbf, T2K * IN_F);
  f2bf_kernel<<<2048, 256, 0, stream>>>(t2o_w,  t2obf, T2N * T2K);

  // 2) ragged gather + concat, gather indices
  build_emb_kernel<<<TT, 256, 0, stream>>>(lhs, cls, eids, tids, emb);
  gind_kernel<<<(TT + 255) / 256, 256, 0, stream>>>(targets, gH, g1, g2);

  // 3) tail projections (WMMA)
  gemm_proj_kernel<<<dim3(TT / 32, T1K / 32), 32, 0, stream>>>(emb, IN_F, t1pbf, T1K, h1);
  gemm_proj_kernel<<<dim3(TT / 32, T2K / 32), 32, 0, stream>>>(emb, IN_F, t2pbf, T2K, h2);

  // 4) fused logits + online logsumexp (WMMA)
  fused_lse_kernel<IN_F, false><<<dim3(TT / 32, sH), 32, 0, stream>>>(emb, hWbf, HEAD_N, head_b, gH, tgtH, pH, bpsH);
  fused_lse_kernel<T1K,  false><<<dim3(TT / 32, s1), 32, 0, stream>>>(h1, t1obf, T1N, nullptr, g1, tgt1, p1, bps1);
  fused_lse_kernel<T2K,  true ><<<dim3(TT / 32, s2), 32, 0, stream>>>(h2, t2obf, T2N, nullptr, g2, tgt2, p2, bps2);

  // 5) per-row composition + loss
  reduce_out_kernel<<<(TT + 255) / 256, 256, 0, stream>>>(targets, tgtH, tgt1, tgt2,
                                                          pH, sH, p1, s1, p2, s2, out);
  loss_kernel<<<1, 256, 0, stream>>>(out, out + TT);
}
